// LocalTransformerEncoder_7078106104053
// MI455X (gfx1250) — matrix operand order repaired and checked by reading, not compiled, not run on hardware
//
#include <hip/hip_runtime.h>
#include <hip/hip_bf16.h>
#include <stdint.h>

#define EMB   512
#define HEADS 8
#define HD    64
#define NTOK  2048
#define INEMB 1024
#define ATT_SCALE 22.627416997969522f   /* sqrt(512) */
#define LN_EPS 1e-5f

typedef __bf16 v16bf __attribute__((ext_vector_type(16)));
typedef __bf16 v8bf  __attribute__((ext_vector_type(8)));
typedef __bf16 v4bf  __attribute__((ext_vector_type(4)));
typedef float  v8f   __attribute__((ext_vector_type(8)));

struct Frag { uint4 lo, hi; };   // 32 bytes == v16bf

__device__ __forceinline__ float wred_sum(float v) {
#pragma unroll
  for (int m = 16; m >= 1; m >>= 1) v += __shfl_xor(v, m, 32);
  return v;
}
__device__ __forceinline__ float wred_max(float v) {
#pragma unroll
  for (int m = 16; m >= 1; m >>= 1) v = fmaxf(v, __shfl_xor(v, m, 32));
  return v;
}

// ---------------------------------------------------------------------------
// C[M,N] = act(A[M,K] @ B[K,N] + bias[N]); bf16 WMMA, f32 accumulate.
// 128 threads = 4 waves (2x2); wave tile 32x32 (2x2 WMMA); block tile 64x64.
// Double-buffered LDS, KB = 32. All addresses hoisted + advanced incrementally.
// ---------------------------------------------------------------------------
__global__ __launch_bounds__(128)
void gemm_bias_act(const float* __restrict__ A, const float* __restrict__ B,
                   const float* __restrict__ bias, float* __restrict__ C,
                   int M, int N, int K, int relu)
{
  __shared__ __align__(16) __bf16 As[2][64][40];   // [row][k], pitch 80B
  __shared__ __align__(16) __bf16 Bs[2][64][40];   // transposed: [col][k]
  const int BUF = 64 * 40;                         // buffer stride (elems)

  const int tid  = threadIdx.x;
  const int lane = tid & 31;
  const int wv   = tid >> 5;
  const int wvr  = wv >> 1;        // 0..1 (M)
  const int wvc  = wv & 1;         // 0..1 (N)
  const int m0   = blockIdx.y * 64;
  const int n0   = blockIdx.x * 64;

  // CDNA5 16-bit WMMA fragment addressing
  const int arow  = lane & 15;
  const int kbase = (lane < 16) ? 0 : 8;     // A lane holds K {kbase..+7, 16+kbase..+7}
  const int khalf = (lane < 16) ? 0 : 16;    // unused; B lane holds the same K as A (kbase)

  // ---- hoisted per-thread addresses ----
  const float* aptr[2];            // global A sources (advance +32 per K-step)
  __bf16*      adst[2];            // LDS dests (buffer 0)
#pragma unroll
  for (int i = 0; i < 2; ++i) {
    int q = tid + 128 * i;
    int r = q >> 2, c8 = (q & 3) << 3;
    aptr[i] = A + (size_t)(m0 + r) * K + c8;
    adst[i] = &As[0][r][c8];
  }
  const float* bptr[4];            // global B sources (advance +32*N per K-step)
  __bf16*      bdst[4];
#pragma unroll
  for (int i = 0; i < 4; ++i) {
    int q = tid + 128 * i;
    int col = q & 63, kk = (q >> 6) << 2;
    bptr[i] = B + (size_t)kk * N + (n0 + col);
    bdst[i] = &Bs[0][col][kk];
  }
  const size_t bstep = (size_t)32 * N;

  const __bf16* afp[2];            // LDS fragment sources (buffer 0)
  const __bf16* bfp[2];
#pragma unroll
  for (int t = 0; t < 2; ++t) {
    afp[t] = &As[0][wvr * 32 + t * 16 + arow][0];
    bfp[t] = &Bs[0][wvc * 32 + t * 16 + arow][0];
  }

  v8f acc[2][2];
#pragma unroll
  for (int i = 0; i < 2; ++i)
#pragma unroll
    for (int j = 0; j < 2; ++j)
      acc[i][j] = (v8f){0.f,0.f,0.f,0.f,0.f,0.f,0.f,0.f};

  // ---- staging: batch loads -> convert -> packed LDS stores ----
  auto stage = [&](int buf) {
    float4 ar[2][2];
#pragma unroll
    for (int i = 0; i < 2; ++i) {
      ar[i][0] = *(const float4*)(aptr[i] + 0);
      ar[i][1] = *(const float4*)(aptr[i] + 4);
      aptr[i] += 32;
    }
    float br[4][4];
#pragma unroll
    for (int i = 0; i < 4; ++i) {
#pragma unroll
      for (int j = 0; j < 4; ++j) br[i][j] = bptr[i][(size_t)j * N];
      bptr[i] += bstep;
    }
    __builtin_prefetch(aptr[0], 0, 1);             // global_prefetch_b8
    __builtin_prefetch(bptr[0], 0, 1);
#pragma unroll
    for (int i = 0; i < 2; ++i) {
      v8bf p;
      p[0] = (__bf16)ar[i][0].x; p[1] = (__bf16)ar[i][0].y;
      p[2] = (__bf16)ar[i][0].z; p[3] = (__bf16)ar[i][0].w;
      p[4] = (__bf16)ar[i][1].x; p[5] = (__bf16)ar[i][1].y;
      p[6] = (__bf16)ar[i][1].z; p[7] = (__bf16)ar[i][1].w;
      *(v8bf*)(adst[i] + buf * BUF) = p;           // ds_store_b128
    }
#pragma unroll
    for (int i = 0; i < 4; ++i) {
      v4bf p;
      p[0] = (__bf16)br[i][0]; p[1] = (__bf16)br[i][1];
      p[2] = (__bf16)br[i][2]; p[3] = (__bf16)br[i][3];
      *(v4bf*)(bdst[i] + buf * BUF) = p;           // ds_store_b64
    }
  };

  // ---- compute one K-step: 2 A-frags x 2 B-frags -> 4 WMMAs ----
  auto compute = [&](int buf) {
    v16bf av[2], bv[2];
#pragma unroll
    for (int t = 0; t < 2; ++t) {
      const __bf16* ap = afp[t] + buf * BUF;
      Frag fa;
      fa.lo = *(const uint4*)(ap + kbase);
      fa.hi = *(const uint4*)(ap + 16 + kbase);
      av[t] = __builtin_bit_cast(v16bf, fa);
      const __bf16* bp = bfp[t] + buf * BUF;
      Frag fb;
      fb.lo = *(const uint4*)(bp + kbase);
      fb.hi = *(const uint4*)(bp + 16 + kbase);
      bv[t] = __builtin_bit_cast(v16bf, fb);
    }
#pragma unroll
    for (int tr = 0; tr < 2; ++tr)
#pragma unroll
      for (int tc = 0; tc < 2; ++tc)
        acc[tr][tc] = __builtin_amdgcn_wmma_f32_16x16x32_bf16(
            false, av[tr], false, bv[tc], (short)0, acc[tr][tc], false, false);
  };

  const int nk = K >> 5;
  stage(0);
  for (int s = 0; s < nk; ++s) {
    __syncthreads();
    if (s + 1 < nk) stage((s + 1) & 1);
    compute(s & 1);
  }

  // ---- epilogue: bias (+relu), fp32 store ----
#pragma unroll
  for (int tr = 0; tr < 2; ++tr) {
#pragma unroll
    for (int tc = 0; tc < 2; ++tc) {
      const int cn = n0 + wvc * 32 + tc * 16 + (lane & 15);
      const int rb = m0 + wvr * 32 + tr * 16 + ((lane >> 4) << 3);
      const float bn = bias[cn];
#pragma unroll
      for (int r = 0; r < 8; ++r) {
        float v = acc[tr][tc][r] + bn;
        if (relu) v = fmaxf(v, 0.f);
        C[(size_t)(rb + r) * N + cn] = v;
      }
    }
  }
}

// ---------------------------------------------------------------------------
// Gathered local attention. One wave per (token, head). KNB = 16 or 64.
// Q/K/V are [NTOK, EMB] with col = h*64 + d. idx is [NTOK, 64].
// ---------------------------------------------------------------------------
template <int KNB>
__global__ __launch_bounds__(256)
void attn_kernel(const float* __restrict__ Q, const float* __restrict__ Kb,
                 const float* __restrict__ Vb, const int* __restrict__ idx,
                 float* __restrict__ Out, float* __restrict__ attsum)
{
  __shared__ __align__(16) float qs[8][64];
  const int tid = threadIdx.x, lane = tid & 31, wv = tid >> 5;
  const int w = blockIdx.x * 8 + wv;
  const int n = w >> 3;          // token
  const int h = w & 7;           // head

  const float* qrow = Q + (size_t)n * EMB + h * HD;
  qs[wv][lane]      = qrow[lane];
  qs[wv][lane + 32] = qrow[lane + 32];
  __syncthreads();

  const int j0   = (KNB == 64) ? lane : (lane & 15);
  const int kix0 = idx[n * 64 + j0];
  const int kix1 = (KNB == 64) ? idx[n * 64 + lane + 32] : 0;

  // energies: q . K[gathered]
  const float4* q4  = (const float4*)qs[wv];
  const float4* kr0 = (const float4*)(Kb + (size_t)kix0 * EMB + h * HD);
  const float4* kr1 = (const float4*)(Kb + (size_t)kix1 * EMB + h * HD);
  float e0 = 0.f, e1 = 0.f;
#pragma unroll
  for (int t = 0; t < 16; ++t) {
    float4 qq = q4[t];
    float4 a  = kr0[t];
    e0 += qq.x * a.x + qq.y * a.y + qq.z * a.z + qq.w * a.w;
    if (KNB == 64) {
      float4 b = kr1[t];
      e1 += qq.x * b.x + qq.y * b.y + qq.z * b.z + qq.w * b.w;
    }
  }

  // softmax over neighbors within the wave, then /SCALE
  float m = (KNB == 64) ? fmaxf(e0, e1) : e0;
  m = wred_max(m);
  float p0 = __expf(e0 - m);
  float p1 = (KNB == 64) ? __expf(e1 - m) : 0.f;
  float s  = wred_sum(p0 + p1);
  if (KNB == 16) s *= 0.5f;            // lanes 16..31 duplicate lanes 0..15 exactly
  const float inv = 1.f / (s * ATT_SCALE);
  const float a0 = p0 * inv, a1 = p1 * inv;

  if (KNB == 64) {                     // patch_att contribution (layer 2 only)
    float tot = wred_sum(a0 + a1);
    if (lane == 0) attsum[h * NTOK + n] = tot;
  }

  // out[d] = sum_j att_j * V[idx_j][d]; lanes cover d = lane, lane+32
  float o0 = 0.f, o1 = 0.f;
  if (KNB == 16) {
#pragma unroll
    for (int j = 0; j < 16; ++j) {
      float aj = __shfl(a0, j, 32);
      int   kj = __shfl(kix0, j, 32);
      const float* vr = Vb + (size_t)kj * EMB + h * HD;
      o0 += aj * vr[lane];
      o1 += aj * vr[lane + 32];
    }
  } else {
#pragma unroll 8
    for (int j = 0; j < 32; ++j) {
      float aj = __shfl(a0, j, 32);
      int   kj = __shfl(kix0, j, 32);
      const float* vr = Vb + (size_t)kj * EMB + h * HD;
      o0 += aj * vr[lane];
      o1 += aj * vr[lane + 32];
      float aj2 = __shfl(a1, j, 32);
      int   kj2 = __shfl(kix1, j, 32);
      const float* vr2 = Vb + (size_t)kj2 * EMB + h * HD;
      o0 += aj2 * vr2[lane];
      o1 += aj2 * vr2[lane + 32];
    }
  }
  float* orow = Out + (size_t)n * EMB + h * HD;
  orow[lane]      = o0;
  orow[lane + 32] = o1;
}

// ---------------------------------------------------------------------------
// Y = LayerNorm(X + R) * g + b ; one wave per token (512 elems, 16/lane).
// ---------------------------------------------------------------------------
__global__ __launch_bounds__(256)
void add_layernorm(const float* __restrict__ X, const float* __restrict__ R,
                   const float* __restrict__ g, const float* __restrict__ b,
                   float* __restrict__ Y)
{
  const int tid = threadIdx.x, lane = tid & 31, wv = tid >> 5;
  const int n = blockIdx.x * 8 + wv;
  const float* xr = X + (size_t)n * EMB;
  const float* rr = R + (size_t)n * EMB;

  float v[16];
  float s = 0.f;
#pragma unroll
  for (int t = 0; t < 16; ++t) {
    int d = lane + 32 * t;
    v[t] = xr[d] + rr[d];
    s += v[t];
  }
  const float mean = wred_sum(s) * (1.f / EMB);

  float var = 0.f;
#pragma unroll
  for (int t = 0; t < 16; ++t) {
    float d = v[t] - mean;
    var += d * d;
  }
  var = wred_sum(var) * (1.f / EMB);
  const float inv = rsqrtf(var + LN_EPS);

  float* yr = Y + (size_t)n * EMB;
#pragma unroll
  for (int t = 0; t < 16; ++t) {
    int d = lane + 32 * t;
    yr[d] = (v[t] - mean) * inv * g[d] + b[d];
  }
}

// Att2[n] = sum_h attsum[h][n]
__global__ __launch_bounds__(256)
void attsum_reduce(const float* __restrict__ attsum, float* __restrict__ out)
{
  int n = blockIdx.x * blockDim.x + threadIdx.x;
  if (n < NTOK) {
    float s = 0.f;
#pragma unroll
    for (int h = 0; h < HEADS; ++h) s += attsum[h * NTOK + n];
    out[n] = s;
  }
}

// ---------------------------------------------------------------------------
extern "C" void kernel_launch(void* const* d_in, const int* in_sizes, int n_in,
                              void* d_out, int out_size, void* d_ws, size_t ws_size,
                              hipStream_t stream)
{
  const float* x    = (const float*)d_in[0];
  const int*   A    = (const int*)  d_in[1];
  const float* W_fc = (const float*)d_in[2];
  const float* b_fc = (const float*)d_in[3];
  const float* Wq1  = (const float*)d_in[4];
  const float* bq1  = (const float*)d_in[5];
  const float* Wk1  = (const float*)d_in[6];
  const float* bk1  = (const float*)d_in[7];
  const float* Wv1  = (const float*)d_in[8];
  const float* bv1  = (const float*)d_in[9];
  const float* g1   = (const float*)d_in[10];
  const float* be1  = (const float*)d_in[11];
  const float* Wq2  = (const float*)d_in[12];
  const float* bq2  = (const float*)d_in[13];
  const float* Wk2  = (const float*)d_in[14];
  const float* bk2  = (const float*)d_in[15];
  const float* Wv2  = (const float*)d_in[16];
  const float* bv2  = (const float*)d_in[17];
  const float* g2   = (const float*)d_in[18];
  const float* be2  = (const float*)d_in[19];

  float* out = (float*)d_out;                 // [NTOK*EMB] b2, then [NTOK] Att2
  char*  ws  = (char*)d_ws;
  const size_t SZ = (size_t)NTOK * EMB * sizeof(float);   // 4 MB each
  float* embs = (float*)(ws + 0 * SZ);
  float* qb   = (float*)(ws + 1 * SZ);
  float* kb   = (float*)(ws + 2 * SZ);
  float* vb   = (float*)(ws + 3 * SZ);
  float* attn = (float*)(ws + 4 * SZ);
  float* b1   = (float*)(ws + 5 * SZ);
  float* asum = (float*)(ws + 6 * SZ);        // HEADS*NTOK floats

  dim3 gblk(128);
  dim3 gg(EMB / 64, NTOK / 64);               // (8, 32) GEMM tile grid
  dim3 blk(256);

  // embs = relu(x @ W_fc + b_fc)
  gemm_bias_act<<<gg, gblk, 0, stream>>>(x, W_fc, b_fc, embs, NTOK, EMB, INEMB, 1);

  // layer 1 projections
  gemm_bias_act<<<gg, gblk, 0, stream>>>(embs, Wq1, bq1, qb, NTOK, EMB, EMB, 0);
  gemm_bias_act<<<gg, gblk, 0, stream>>>(embs, Wk1, bk1, kb, NTOK, EMB, EMB, 0);
  gemm_bias_act<<<gg, gblk, 0, stream>>>(embs, Wv1, bv1, vb, NTOK, EMB, EMB, 0);
  attn_kernel<16><<<NTOK * HEADS / 8, blk, 0, stream>>>(qb, kb, vb, A, attn, nullptr);
  add_layernorm<<<NTOK / 8, blk, 0, stream>>>(attn, embs, g1, be1, b1);

  // layer 2 projections
  gemm_bias_act<<<gg, gblk, 0, stream>>>(b1, Wq2, bq2, qb, NTOK, EMB, EMB, 0);
  gemm_bias_act<<<gg, gblk, 0, stream>>>(b1, Wk2, bk2, kb, NTOK, EMB, EMB, 0);
  gemm_bias_act<<<gg, gblk, 0, stream>>>(b1, Wv2, bv2, vb, NTOK, EMB, EMB, 0);
  attn_kernel<64><<<NTOK * HEADS / 8, blk, 0, stream>>>(qb, kb, vb, A, attn, asum);
  add_layernorm<<<NTOK / 8, blk, 0, stream>>>(attn, b1, g2, be2, out);

  attsum_reduce<<<NTOK / 256, 256, 0, stream>>>(asum, out + (size_t)NTOK * EMB);
}
